// Block_69784628625493
// MI455X (gfx1250) — matrix operand order repaired and checked by reading, not compiled, not run on hardware
//
#include <hip/hip_runtime.h>
#include <hip/hip_bf16.h>

// ---------------- problem constants (from reference) ----------------
#define B_   4
#define TT_  2048
#define C_   2048
#define H_   32
#define S_   64
#define DF_  7168
#define TC_  512
#define NC_  4
static constexpr float EPSV   = 1e-5f;
static constexpr float HS_DIV = 8.0f;   // sqrt(S)

typedef __bf16 bf16_t;
typedef __attribute__((ext_vector_type(16))) __bf16 v16bf;
typedef __attribute__((ext_vector_type(8)))  float  v8f;
typedef __attribute__((ext_vector_type(4)))  int    v4i;

union FragBF { v16bf v; unsigned int u[8]; __bf16 e[16]; };
union PairBF { __bf16 h[2]; unsigned int u; };

__device__ __forceinline__ v8f vzero() {
    v8f z = {0.f,0.f,0.f,0.f,0.f,0.f,0.f,0.f};
    return z;
}
__device__ __forceinline__ v8f wmma_bf16(v16bf a, v16bf b, v8f c) {
    return __builtin_amdgcn_wmma_f32_16x16x32_bf16(false, a, false, b, (short)0, c, false, false);
}
// A-frag (16x32 bf16) K offset for vgpr v and lane-half h (contiguous: k = 16h + 2v)
__device__ __forceinline__ int a_koff(int v, int half) {
    return half * 16 + v * 2;
}

// ---------------- async global->LDS copy (gfx1250 ASYNCcnt path) ----------------
#if defined(__has_builtin)
#if __has_builtin(__builtin_amdgcn_global_load_async_to_lds_b128) && \
    __has_builtin(__builtin_amdgcn_s_wait_asynccnt)
#define USE_ASYNC_LDS 1
#endif
#endif
#ifndef USE_ASYNC_LDS
#define USE_ASYNC_LDS 0
#endif

__device__ __forceinline__ void async_copy16(const bf16_t* g, bf16_t* l) {
#if USE_ASYNC_LDS
    __builtin_amdgcn_global_load_async_to_lds_b128(
        (__attribute__((address_space(1))) v4i*)g,
        (__attribute__((address_space(3))) v4i*)l, 0, 0);
#else
    *(uint4*)l = *(const uint4*)g;
#endif
}
__device__ __forceinline__ void async_wait() {
#if USE_ASYNC_LDS
    __builtin_amdgcn_s_wait_asynccnt(0);
#endif
}

// ---------------- weight conversion f32 -> bf16 ----------------
__global__ void f32_to_bf16(const float* __restrict__ in, bf16_t* __restrict__ out, int n) {
    int i = blockIdx.x * blockDim.x + threadIdx.x;
    int stride = gridDim.x * blockDim.x;
    for (; i < n; i += stride) out[i] = (bf16_t)in[i];
}

// ---------------- fused LayerNorm + time-shift + mix ----------------
// One block per token row. Recomputes LN of the previous row (t-1) so blocks
// are independent; t==0 rows use zero for the shifted term (pad semantics).
template<int NOUT>
__global__ __launch_bounds__(256, 1)
void ln_mix(const float* __restrict__ x, const float* __restrict__ g,
            const float* __restrict__ bb,
            const float* __restrict__ mk, const float* __restrict__ mv,
            const float* __restrict__ mr,
            bf16_t* __restrict__ ok, bf16_t* __restrict__ ov,
            bf16_t* __restrict__ orr)
{
    __shared__ float red[4][8];
    const int row = blockIdx.x;
    const int t   = row % TT_;
    const int tid = threadIdx.x;
    const float* xc = x + (size_t)row * C_;
    const float* xp = xc - C_;

    float cv[8], pv[8];
    float sc = 0.f, sc2 = 0.f, sp = 0.f, sp2 = 0.f;
#pragma unroll
    for (int i = 0; i < 8; ++i) {
        int idx = i * 256 + tid;
        float a = xc[idx];
        cv[i] = a; sc += a; sc2 += a * a;
        float p = (t > 0) ? xp[idx] : 0.f;
        pv[i] = p; sp += p; sp2 += p * p;
    }
#pragma unroll
    for (int o = 16; o > 0; o >>= 1) {
        sc  += __shfl_xor(sc,  o, 32);
        sc2 += __shfl_xor(sc2, o, 32);
        sp  += __shfl_xor(sp,  o, 32);
        sp2 += __shfl_xor(sp2, o, 32);
    }
    if ((tid & 31) == 0) {
        int w = tid >> 5;
        red[0][w] = sc; red[1][w] = sc2; red[2][w] = sp; red[3][w] = sp2;
    }
    __syncthreads();
    float tc = 0.f, tc2 = 0.f, tp = 0.f, tp2 = 0.f;
#pragma unroll
    for (int w = 0; w < 8; ++w) {
        tc += red[0][w]; tc2 += red[1][w]; tp += red[2][w]; tp2 += red[3][w];
    }
    const float invC = 1.0f / (float)C_;
    float mc = tc * invC, vc = tc2 * invC - mc * mc;
    float mp = tp * invC, vp = tp2 * invC - mp * mp;
    float ic = rsqrtf(vc + EPSV);
    float ip = rsqrtf(vp + EPSV);

#pragma unroll
    for (int i = 0; i < 8; ++i) {
        int idx = i * 256 + tid;
        float gi = g[idx], bi = bb[idx];
        float h1c = (cv[i] - mc) * ic * gi + bi;
        float h1p = (t > 0) ? ((pv[i] - mp) * ip * gi + bi) : 0.f;
        size_t o = (size_t)row * C_ + idx;
        float a = mk[idx];
        ok[o] = (bf16_t)(h1c * a + h1p * (1.f - a));
        if (NOUT == 3) {
            float b2 = mv[idx];
            ov[o] = (bf16_t)(h1c * b2 + h1p * (1.f - b2));
        }
        float r2 = mr[idx];
        orr[o] = (bf16_t)(h1c * r2 + h1p * (1.f - r2));
    }
}

// ---------------- bf16 WMMA GEMM, 128x256 block tile ----------------
// 256 threads = 8 waves (2x4); wave tile 64x64 (4x4 wmma tiles), K-step 32.
// A: (M,K) row-major bf16, staged via async global->LDS.
// W: (K,N) row-major bf16, staged transposed into Bsh[n][k] as (k,k+1) pairs.
// Epilogues: 0 store bf16 | 1 res + acc -> f32 | 2 relu^2 -> bf16
//            3 sigmoid -> f32            | 4 res + mul*acc -> f32
template<int EPI>
__global__ __launch_bounds__(256, 1)
void gemm_bf16(const bf16_t* __restrict__ A, const bf16_t* __restrict__ W,
               float* outf, bf16_t* outb, const float* res, const float* mul,
               int M, int N, int K)
{
    __shared__ bf16_t Ash[128 * 40];   // [m][k] stride 40
    __shared__ bf16_t Bsh[256 * 40];   // [n][k] stride 40 (transposed)

    const int tid  = threadIdx.x;
    const int lane = tid & 31, wid = tid >> 5;
    const int half = lane >> 4, l16 = lane & 15;
    const int bm = blockIdx.y * 128, bn = blockIdx.x * 256;
    const int wm = (wid >> 2) * 64, wn = (wid & 3) * 64;

    v8f acc[4][4];
#pragma unroll
    for (int i = 0; i < 4; ++i)
#pragma unroll
        for (int j = 0; j < 4; ++j) acc[i][j] = vzero();

    const int ar = tid >> 1, ak = (tid & 1) * 16;   // A staging: 16 bf16/thread
    const int kp = (tid >> 4) * 2;                  // B staging: 2 k-rows ...
    const int n0 = (tid & 15) * 16;                 // ... x 16 n-cols /thread

    for (int k0 = 0; k0 < K; k0 += 32) {
        __syncthreads();
        const bf16_t* asrc = A + (size_t)(bm + ar) * K + k0 + ak;
        const bf16_t* bs0  = W + (size_t)(k0 + kp) * N + bn + n0;
        const bf16_t* bs1  = bs0 + N;
        if (k0 + 32 < K) {  // prefetch next K-tile of W (global_prefetch_b8)
            __builtin_prefetch(bs0 + (size_t)32 * N, 0, 1);
            __builtin_prefetch(bs1 + (size_t)32 * N, 0, 1);
        }
        // stage A 128x32 (async -> LDS)
        async_copy16(asrc,     &Ash[ar * 40 + ak]);
        async_copy16(asrc + 8, &Ash[ar * 40 + ak + 8]);
        {   // stage B 32x256 transposed, packed as (k,k+1) pairs
            bf16_t r0[16], r1[16];
            *(uint4*)(r0)     = *(const uint4*)(bs0);
            *(uint4*)(r0 + 8) = *(const uint4*)(bs0 + 8);
            *(uint4*)(r1)     = *(const uint4*)(bs1);
            *(uint4*)(r1 + 8) = *(const uint4*)(bs1 + 8);
#pragma unroll
            for (int j = 0; j < 16; ++j) {
                PairBF p; p.h[0] = r0[j]; p.h[1] = r1[j];
                *(unsigned int*)(&Bsh[(n0 + j) * 40 + kp]) = p.u;
            }
        }
        async_wait();
        __syncthreads();

        v16bf afr[4];
#pragma unroll
        for (int mt = 0; mt < 4; ++mt) {
            int m = wm + mt * 16 + l16;
            FragBF f;
#pragma unroll
            for (int v = 0; v < 8; ++v)
                f.u[v] = *(const unsigned int*)(&Ash[m * 40 + a_koff(v, half)]);
            afr[mt] = f.v;
        }
#pragma unroll
        for (int nt = 0; nt < 4; ++nt) {
            int n = wn + nt * 16 + l16;
            FragBF f;
#pragma unroll
            for (int v = 0; v < 8; ++v)
                f.u[v] = *(const unsigned int*)(&Bsh[n * 40 + half * 16 + 2 * v]);
#pragma unroll
            for (int mt = 0; mt < 4; ++mt)
                acc[mt][nt] = wmma_bf16(afr[mt], f.v, acc[mt][nt]);
        }
    }

    // epilogue (C/D layout: row = 8*half + g, col = l16)
#pragma unroll
    for (int mt = 0; mt < 4; ++mt)
#pragma unroll
        for (int nt = 0; nt < 4; ++nt)
#pragma unroll
            for (int g = 0; g < 8; ++g) {
                int r = wm + mt * 16 + half * 8 + g;
                int c = wn + nt * 16 + l16;
                size_t idx = (size_t)(bm + r) * N + (bn + c);
                float a = acc[mt][nt][g];
                if (EPI == 0) outb[idx] = (bf16_t)a;
                else if (EPI == 1) outf[idx] = res[idx] + a;
                else if (EPI == 2) { float t = a > 0.f ? a : 0.f; outb[idx] = (bf16_t)(t * t); }
                else if (EPI == 3) outf[idx] = 1.f / (1.f + __expf(-a));
                else               outf[idx] = res[idx] + mul[idx] * a;
            }
}

// ---------------- WKV chunked linear attention (per (b,h)) ----------------
// One workgroup (8 waves) per (b,h). State s (64x64) lives in LDS as bf16
// s^T; chunks processed sequentially. Each wave owns 64 t-rows of the chunk.
__global__ __launch_bounds__(256, 1)
void wkv_attn(const bf16_t* __restrict__ Rg, const bf16_t* __restrict__ Kg,
              const bf16_t* __restrict__ Vg, const float* __restrict__ decay,
              const float* __restrict__ faaaa, float* __restrict__ Xout)
{
    __shared__ bf16_t sT[64 * 66];        // sT[j*66+i] = state[i][j]
    __shared__ bf16_t ksh[64 * 72];       // ksh[u*72+s] = k[t0+u][s]
    __shared__ bf16_t vT[64 * 72];        // vT[s*72+u]  = v[t0+u][s]
    __shared__ bf16_t ash[8 * 64 * 34];   // per-wave masked a (64t x 32u)
    __shared__ float  wpow[TC_];

    const int bh = blockIdx.x, b = bh / H_, h = bh % H_;
    const int tid = threadIdx.x, lane = tid & 31, wid = tid >> 5;
    const int half = lane >> 4, l16 = lane & 15;

    const float nlw   = -__expf(decay[h]);     // log(w) = -exp(decay)
    const float ucoef = faaaa[h];
    const float ws    = __expf(nlw * (float)TC_);

    for (int e = tid; e < TC_; e += 256) wpow[e] = __expf(nlw * (float)e);
    for (int i = tid; i < 64 * 66; i += 256) sT[i] = (bf16_t)0.f;
    __syncthreads();

    const size_t rowbase = (size_t)b * TT_ * C_ + (size_t)h * S_;

    for (int c = 0; c < NC_; ++c) {
        const size_t crow = (size_t)c * TC_;

        // r fragments for this wave's 64 rows: 4 m-tiles x 2 k-steps
        v16bf rfr[4][2];
#pragma unroll
        for (int mt = 0; mt < 4; ++mt) {
            int trow = wid * 64 + mt * 16 + l16;
            const bf16_t* rp = Rg + rowbase + (crow + trow) * C_;
#pragma unroll
            for (int ks = 0; ks < 2; ++ks) {
                FragBF f;
#pragma unroll
                for (int v = 0; v < 8; ++v)
                    f.u[v] = *(const unsigned int*)(rp + ks * 32 + a_koff(v, half));
                rfr[mt][ks] = f.v;
            }
        }

        // out init: (r @ s_old) scaled per-row by w^t
        v8f oacc[4][4];
#pragma unroll
        for (int mt = 0; mt < 4; ++mt)
#pragma unroll
            for (int nt = 0; nt < 4; ++nt) {
                v8f t = vzero();
#pragma unroll
                for (int ks = 0; ks < 2; ++ks) {
                    FragBF f;
#pragma unroll
                    for (int v = 0; v < 8; ++v)
                        f.u[v] = *(const unsigned int*)(
                            &sT[(nt * 16 + l16) * 66 + ks * 32 + half * 16 + 2 * v]);
                    t = wmma_bf16(rfr[mt][ks], f.v, t);
                }
#pragma unroll
                for (int g = 0; g < 8; ++g)
                    t[g] *= wpow[wid * 64 + mt * 16 + half * 8 + g];
                oacc[mt][nt] = t;
            }

        v8f sacc[2] = { vzero(), vzero() };   // this wave's 2 state tiles

        for (int ub = 0; ub < 8; ++ub) {
            __syncthreads();
            {   // stage k (row-major, async) and v (transposed) for 64 u values
                int u = tid >> 2, sp = (tid & 3) * 16;
                const bf16_t* kp = Kg + rowbase + (crow + ub * 64 + u) * C_ + sp;
                const bf16_t* vp = Vg + rowbase + (crow + ub * 64 + u) * C_ + sp;
                async_copy16(kp,     &ksh[u * 72 + sp]);
                async_copy16(kp + 8, &ksh[u * 72 + sp + 8]);
                bf16_t tmp[16];
                *(uint4*)(tmp)     = *(const uint4*)(vp);
                *(uint4*)(tmp + 8) = *(const uint4*)(vp + 8);
#pragma unroll
                for (int j = 0; j < 16; ++j) vT[(sp + j) * 72 + u] = tmp[j];
            }
            async_wait();
            __syncthreads();

            for (int us = 0; us < 2; ++us) {
                // P = r @ k^T over 32 u values
                v8f pacc[4][2];
#pragma unroll
                for (int mt = 0; mt < 4; ++mt)
#pragma unroll
                    for (int nt = 0; nt < 2; ++nt) {
                        v8f t = vzero();
#pragma unroll
                        for (int ks = 0; ks < 2; ++ks) {
                            FragBF f;
#pragma unroll
                            for (int v = 0; v < 8; ++v)
                                f.u[v] = *(const unsigned int*)(
                                    &ksh[(us * 32 + nt * 16 + l16) * 72 +
                                         ks * 32 + half * 16 + 2 * v]);
                            t = wmma_bf16(rfr[mt][ks], f.v, t);
                        }
                        pacc[mt][nt] = t;
                    }
                // decay mask and spill to per-wave LDS as bf16
#pragma unroll
                for (int mt = 0; mt < 4; ++mt)
#pragma unroll
                    for (int nt = 0; nt < 2; ++nt)
#pragma unroll
                        for (int g = 0; g < 8; ++g) {
                            int tl = wid * 64 + mt * 16 + half * 8 + g;
                            int ul = ub * 64 + us * 32 + nt * 16 + l16;
                            float m = (ul < tl) ? wpow[tl - ul - 1]
                                                : ((ul == tl) ? ucoef : 0.f);
                            ash[wid * 2176 + (mt * 16 + half * 8 + g) * 34 +
                                nt * 16 + l16] = (bf16_t)(pacc[mt][nt][g] * m);
                        }
                __syncthreads();   // cross-lane LDS RAW
                // out += a @ v
                v16bf afr[4];
#pragma unroll
                for (int mt = 0; mt < 4; ++mt) {
                    FragBF f;
#pragma unroll
                    for (int v = 0; v < 8; ++v)
                        f.u[v] = *(const unsigned int*)(
                            &ash[wid * 2176 + (mt * 16 + l16) * 34 + a_koff(v, half)]);
                    afr[mt] = f.v;
                }
#pragma unroll
                for (int nt = 0; nt < 4; ++nt) {
                    FragBF f;
#pragma unroll
                    for (int v = 0; v < 8; ++v)
                        f.u[v] = *(const unsigned int*)(
                            &vT[(nt * 16 + l16) * 72 + us * 32 + half * 16 + 2 * v]);
#pragma unroll
                    for (int mt = 0; mt < 4; ++mt)
                        oacc[mt][nt] = wmma_bf16(afr[mt], f.v, oacc[mt][nt]);
                }
                __syncthreads();   // protect ash before next us overwrite
            }

            // state update partial: sacc += (k * w^(TC-1-t))^T @ v
#pragma unroll
            for (int tt = 0; tt < 2; ++tt) {
                int ti = (wid * 2 + tt) >> 2, tj = (wid * 2 + tt) & 3;
#pragma unroll
                for (int ks = 0; ks < 2; ++ks) {
                    int t0 = ks * 32;
                    FragBF fa;
#pragma unroll
                    for (int v = 0; v < 8; ++v) {
                        int toff = t0 + a_koff(v, half);
                        int tg = ub * 64 + toff;
                        float k0 = (float)ksh[toff * 72 + ti * 16 + l16] * wpow[TC_ - 1 - tg];
                        float k1 = (float)ksh[(toff + 1) * 72 + ti * 16 + l16] * wpow[TC_ - 2 - tg];
                        fa.e[2 * v]     = (bf16_t)k0;
                        fa.e[2 * v + 1] = (bf16_t)k1;
                    }
                    FragBF fb;
#pragma unroll
                    for (int v = 0; v < 8; ++v)
                        fb.u[v] = *(const unsigned int*)(
                            &vT[(tj * 16 + l16) * 72 + t0 + half * 16 + 2 * v]);
                    sacc[tt] = wmma_bf16(fa.v, fb.v, sacc[tt]);
                }
            }
        } // ub

        __syncthreads();
        // s_new = ws * s_old + sacc  (each wave owns 2 disjoint 16x16 tiles)
#pragma unroll
        for (int tt = 0; tt < 2; ++tt) {
            int ti = (wid * 2 + tt) >> 2, tj = (wid * 2 + tt) & 3;
#pragma unroll
            for (int g = 0; g < 8; ++g) {
                int i = ti * 16 + half * 8 + g;
                int j = tj * 16 + l16;
                float old = (float)sT[j * 66 + i];
                sT[j * 66 + i] = (bf16_t)(ws * old + sacc[tt][g]);
            }
        }
        // write this chunk's output rows (fp32, (B*TT, C) layout)
#pragma unroll
        for (int mt = 0; mt < 4; ++mt)
#pragma unroll
            for (int nt = 0; nt < 4; ++nt)
#pragma unroll
                for (int g = 0; g < 8; ++g) {
                    int tl = wid * 64 + mt * 16 + half * 8 + g;
                    int col = nt * 16 + l16;
                    Xout[(size_t)(b * TT_ + c * TC_ + tl) * C_ + h * S_ + col] =
                        oacc[mt][nt][g];
                }
        __syncthreads();
    } // chunk
}

// ---------------- per-head group norm (lnx), includes /sqrt(S) ----------------
__global__ __launch_bounds__(256, 1)
void gnorm(const float* __restrict__ xin, const float* __restrict__ gg,
           const float* __restrict__ bb, bf16_t* __restrict__ outb)
{
    const int row = blockIdx.x, tid = threadIdx.x;
    const int h = tid >> 3, s0 = (tid & 7) * 8;   // 8 threads/head, 8 elems each
    const float* p = xin + (size_t)row * C_ + h * S_ + s0;
    float v[8], s = 0.f, s2 = 0.f;
#pragma unroll
    for (int i = 0; i < 8; ++i) {
        float a = p[i] * (1.0f / HS_DIV);
        v[i] = a; s += a; s2 += a * a;
    }
#pragma unroll
    for (int o = 1; o < 8; o <<= 1) { s += __shfl_xor(s, o, 32); s2 += __shfl_xor(s2, o, 32); }
    float m = s * (1.0f / S_), var = s2 * (1.0f / S_) - m * m;
    float inv = rsqrtf(var + EPSV);
#pragma unroll
    for (int i = 0; i < 8; ++i) {
        int col = h * S_ + s0 + i;
        outb[(size_t)row * C_ + col] = (bf16_t)((v[i] - m) * inv * gg[col] + bb[col]);
    }
}

// ---------------- launcher ----------------
extern "C" void kernel_launch(void* const* d_in, const int* in_sizes, int n_in,
                              void* d_out, int out_size, void* d_ws, size_t ws_size,
                              hipStream_t stream)
{
    (void)in_sizes; (void)n_in; (void)out_size; (void)ws_size;
    const float* x      = (const float*)d_in[0];
    const float* ln1_g  = (const float*)d_in[1];
    const float* ln1_b  = (const float*)d_in[2];
    const float* ln2_g  = (const float*)d_in[3];
    const float* ln2_b  = (const float*)d_in[4];
    const float* amk    = (const float*)d_in[5];
    const float* amv    = (const float*)d_in[6];
    const float* amr    = (const float*)d_in[7];
    const float* decay  = (const float*)d_in[8];
    const float* faaaa  = (const float*)d_in[9];
    const float* Wr     = (const float*)d_in[10];
    const float* Wk     = (const float*)d_in[11];
    const float* Wv     = (const float*)d_in[12];
    const float* Wo     = (const float*)d_in[13];
    const float* lnx_g  = (const float*)d_in[14];
    const float* lnx_b  = (const float*)d_in[15];
    const float* fmk    = (const float*)d_in[16];
    const float* fmr    = (const float*)d_in[17];
    const float* Wfk    = (const float*)d_in[18];
    const float* Wfr    = (const float*)d_in[19];
    const float* Wfv    = (const float*)d_in[20];
    float* out = (float*)d_out;

    const int M = B_ * TT_;
    char* ws = (char*)d_ws;
    size_t off = 0;
    auto alloc = [&](size_t bytes) -> void* {
        void* p = ws + off;
        off += (bytes + 255) & ~(size_t)255;
        return p;
    };
    bf16_t* wr  = (bf16_t*)alloc((size_t)C_ * C_ * 2);
    bf16_t* wk  = (bf16_t*)alloc((size_t)C_ * C_ * 2);
    bf16_t* wv  = (bf16_t*)alloc((size_t)C_ * C_ * 2);
    bf16_t* wo  = (bf16_t*)alloc((size_t)C_ * C_ * 2);
    bf16_t* wfr = (bf16_t*)alloc((size_t)C_ * C_ * 2);
    bf16_t* wfk = (bf16_t*)alloc((size_t)C_ * DF_ * 2);
    bf16_t* wfv = (bf16_t*)alloc((size_t)DF_ * C_ * 2);
    bf16_t* xk  = (bf16_t*)alloc((size_t)M * C_ * 2);
    bf16_t* xv  = (bf16_t*)alloc((size_t)M * C_ * 2);
    bf16_t* xr  = (bf16_t*)alloc((size_t)M * C_ * 2);
    bf16_t* rb  = (bf16_t*)alloc((size_t)M * C_ * 2);
    bf16_t* kb  = (bf16_t*)alloc((size_t)M * C_ * 2);
    bf16_t* vb  = (bf16_t*)alloc((size_t)M * C_ * 2);
    float*  xo  = (float*) alloc((size_t)M * C_ * 4);
    bf16_t* xab = (bf16_t*)alloc((size_t)M * C_ * 2);
    bf16_t* xk2 = (bf16_t*)alloc((size_t)M * C_ * 2);
    bf16_t* xr2 = (bf16_t*)alloc((size_t)M * C_ * 2);
    bf16_t* kfb = (bf16_t*)alloc((size_t)M * DF_ * 2);
    float*  srf = (float*) alloc((size_t)M * C_ * 4);

    // 1. weights -> bf16
    f32_to_bf16<<<2048, 256, 0, stream>>>(Wr,  wr,  C_ * C_);
    f32_to_bf16<<<2048, 256, 0, stream>>>(Wk,  wk,  C_ * C_);
    f32_to_bf16<<<2048, 256, 0, stream>>>(Wv,  wv,  C_ * C_);
    f32_to_bf16<<<2048, 256, 0, stream>>>(Wo,  wo,  C_ * C_);
    f32_to_bf16<<<2048, 256, 0, stream>>>(Wfr, wfr, C_ * C_);
    f32_to_bf16<<<4096, 256, 0, stream>>>(Wfk, wfk, C_ * DF_);
    f32_to_bf16<<<4096, 256, 0, stream>>>(Wfv, wfv, DF_ * C_);

    // 2. LN1 + shift + attention mixes
    ln_mix<3><<<M, 256, 0, stream>>>(x, ln1_g, ln1_b, amk, amv, amr, xk, xv, xr);

    // 3. r, k, v projections
    dim3 gCC(C_ / 256, M / 128);
    gemm_bf16<0><<<gCC, 256, 0, stream>>>(xr, wr, nullptr, rb, nullptr, nullptr, M, C_, C_);
    gemm_bf16<0><<<gCC, 256, 0, stream>>>(xk, wk, nullptr, kb, nullptr, nullptr, M, C_, C_);
    gemm_bf16<0><<<gCC, 256, 0, stream>>>(xv, wv, nullptr, vb, nullptr, nullptr, M, C_, C_);

    // 4. WKV chunked scan
    wkv_attn<<<B_ * H_, 256, 0, stream>>>(rb, kb, vb, decay, faaaa, xo);

    // 5. per-head groupnorm (+ /sqrt(S))
    gnorm<<<M, 256, 0, stream>>>(xo, lnx_g, lnx_b, xab);

    // 6. output projection + residual -> d_out (f32)
    gemm_bf16<1><<<gCC, 256, 0, stream>>>(xab, wo, out, nullptr, x, nullptr, M, C_, C_);

    // 7. LN2 + shift + FFN mixes
    ln_mix<2><<<M, 256, 0, stream>>>(out, ln2_g, ln2_b, fmk, nullptr, fmr,
                                     xk2, nullptr, xr2);

    // 8. kf = relu(xk2 @ Wfk)^2
    dim3 gDF(DF_ / 256, M / 128);
    gemm_bf16<2><<<gDF, 256, 0, stream>>>(xk2, wfk, nullptr, kfb, nullptr, nullptr, M, DF_, C_);

    // 9. sr = sigmoid(xr2 @ Wfr)
    gemm_bf16<3><<<gCC, 256, 0, stream>>>(xr2, wfr, srf, nullptr, nullptr, nullptr, M, C_, C_);

    // 10. out = out + sr * (kf @ Wfv)
    gemm_bf16<4><<<gCC, 256, 0, stream>>>(kfb, wfv, out, nullptr, out, srf, M, C_, DF_);
}
